// GraphConv_69879117906747
// MI455X (gfx1250) — matrix operand order repaired and checked by reading, not compiled, not run on hardware
//
#include <hip/hip_runtime.h>

#define IN_F 128
#define OUT_F 128
#define NB 4
#define NN 4096

typedef __attribute__((ext_vector_type(16))) __bf16 v16bf;
typedef __attribute__((ext_vector_type(8)))  float  v8f;
typedef __attribute__((ext_vector_type(8)))  unsigned int v8u;

__device__ __forceinline__ unsigned int f2bf(float f) {   // RNE (epilogue only)
    unsigned int u = __builtin_bit_cast(unsigned int, f);
    return (u + 0x7FFFu + ((u >> 16) & 1u)) >> 16;
}
__device__ __forceinline__ float bf2f(unsigned int h) {
    return __builtin_bit_cast(float, h << 16);
}

// Split two floats into packed-bf16 hi (truncated) + lo (residual).
// Truncation of hi is lossless overall: dropped bits land exactly in lo.
__device__ __forceinline__ void split2_fast(float x0, float x1,
                                            unsigned int& hp, unsigned int& lp) {
    unsigned int u0 = __builtin_bit_cast(unsigned int, x0);
    unsigned int u1 = __builtin_bit_cast(unsigned int, x1);
    float r0 = x0 - __builtin_bit_cast(float, u0 & 0xFFFF0000u);
    float r1 = x1 - __builtin_bit_cast(float, u1 & 0xFFFF0000u);
    unsigned int v0 = __builtin_bit_cast(unsigned int, r0);
    unsigned int v1 = __builtin_bit_cast(unsigned int, r1);
    hp = __builtin_amdgcn_perm(u1, u0, 0x07060302u);  // {u1.hi16, u0.hi16}
    lp = __builtin_amdgcn_perm(v1, v0, 0x07060302u);
}

__device__ __forceinline__ v8f wmma_bf16(v8u a, v8u b, v8f c) {
    return __builtin_amdgcn_wmma_f32_16x16x32_bf16(
        false, __builtin_bit_cast(v16bf, a),
        false, __builtin_bit_cast(v16bf, b),
        (short)0, c, false, false);
}

// ---- Prep: pack W (128x128 fp32) into bf16 hi/lo B-operand fragments in ws.
// Layout: hi plane ws[0..8191], lo plane ws[8192..16383]; fragment dword index
// ((kt*2 + kh)*128 + n)*8 + r  <->  pair (k = kt*32 + 16*(r>>2) + kh*8 + 2*(r&3), k+1) col n.
__global__ __launch_bounds__(256) void pack_w_kernel(
    const float* __restrict__ W, unsigned int* __restrict__ wsW)
{
    const int t = threadIdx.x;
    #pragma unroll
    for (int i = 0; i < 32; ++i) {
        int p  = t + i * 256;            // 0..8191, coalesced stores
        int r  = p & 7;
        int n  = (p >> 3) & 127;
        int kh = (p >> 10) & 1;
        int kt = p >> 11;
        int k2 = kt * 32 + ((r < 4) ? (kh * 8 + 2 * r) : (16 + kh * 8 + 2 * (r - 4)));
        float w0 = W[(size_t)k2 * OUT_F + n];
        float w1 = W[(size_t)(k2 + 1) * OUT_F + n];
        unsigned int hp, lp; split2_fast(w0, w1, hp, lp);
        int idx = ((kt * 2 + kh) * 128 + n) * 8 + r;
        wsW[idx] = hp;
        wsW[idx + 8192] = lp;
    }
}

// ---- Fused: per block, 128 rows of (A@X + X)/deg then @W + relu.
// 256 threads = 8 wave32; wave w owns rows [w*16, w*16+16) x all 128 features.
__global__ __launch_bounds__(256) void graphconv_fused(
    const float* __restrict__ A, const float* __restrict__ X,
    const unsigned int* __restrict__ wsW, float* __restrict__ out)
{
    __shared__ __align__(32) unsigned int smem[16384];   // 64 KB
    unsigned int* sXh = smem;            // X slice hi, B-frag layout (2048 dw)
    unsigned int* sXl = smem + 2048;     // X slice lo
    // phase 2 aliases all 64 KB as per-wave H tiles.

    const int tid = threadIdx.x;
    const int l   = tid & 31;
    const int w   = tid >> 5;
    const int flatRow0 = blockIdx.x * 128;
    const int b    = flatRow0 >> 12;             // / N
    const int row0 = flatRow0 & (NN - 1);
    const float* Ab = A + (size_t)b * NN * NN;
    const float* Xb = X + (size_t)b * NN * IN_F;

    // X staging: thread owns (khalf = tid>>7, n = tid&127), all 8 r-pairs.
    const int nst = tid & 127;
    const int khs = tid >> 7;
    // A fragments: lane owns (m = w*16 + (l&15), ksel = l>>4), 8 r-pairs.
    const float* Arow = Ab + (size_t)(row0 + w * 16 + (l & 15)) * NN + (l >> 4) * 8;

    v8f acc[8];
    v8f dacc;
    #pragma unroll
    for (int c = 0; c < 8; ++c)
        #pragma unroll
        for (int i = 0; i < 8; ++i) acc[c][i] = 0.0f;
    #pragma unroll
    for (int i = 0; i < 8; ++i) dacc[i] = 0.0f;

    v8u onesu;                                   // bf16 1.0 pairs
    #pragma unroll
    for (int i = 0; i < 8; ++i) onesu[i] = 0x3F803F80u;

    // Register double-buffers (pipeline: load k+32 while computing k).
    float  xg0[8], xg1[8];
    float2 ag[8];
    #pragma unroll
    for (int r = 0; r < 8; ++r) {
        int k = (r < 4) ? (khs * 8 + 2 * r) : (16 + khs * 8 + 2 * (r - 4));
        xg0[r] = Xb[(size_t)k * IN_F + nst];
        xg1[r] = Xb[(size_t)(k + 1) * IN_F + nst];
        int klo = (r < 4) ? (2 * r) : (16 + 2 * (r - 4));
        ag[r] = *(const float2*)(Arow + klo);
    }

    const int xidx = (khs * 128 + nst) * 8;
    const int bfh  = l >> 4;

    // ---------------- Phase 1: acc = A@X via bf16 hi/lo x3; dacc = rowsum(A) --
    for (int kk = 0; kk < NN; kk += 32) {
        v8u hx, lx;
        #pragma unroll
        for (int r = 0; r < 8; ++r) {
            unsigned int hp, lp;
            split2_fast(xg0[r], xg1[r], hp, lp);
            hx[r] = hp; lx[r] = lp;
        }
        v8u ah, al;
        #pragma unroll
        for (int r = 0; r < 8; ++r) {
            unsigned int hp, lp;
            split2_fast(ag[r].x, ag[r].y, hp, lp);
            ah[r] = hp; al[r] = lp;
        }

        __syncthreads();                 // prior iteration's LDS reads done
        *(v8u*)&sXh[xidx] = hx;          // 2x ds_store_b128
        *(v8u*)&sXl[xidx] = lx;
        __syncthreads();                 // X slice visible to all waves

        // Next step's global loads overlap the WMMA block below.
        if (kk + 32 < NN) {
            const int kn = kk + 32;
            #pragma unroll
            for (int r = 0; r < 8; ++r) {
                int k = kn + ((r < 4) ? (khs * 8 + 2 * r) : (16 + khs * 8 + 2 * (r - 4)));
                xg0[r] = Xb[(size_t)k * IN_F + nst];
                xg1[r] = Xb[(size_t)(k + 1) * IN_F + nst];
                int klo = (r < 4) ? (2 * r) : (16 + 2 * (r - 4));
                ag[r] = *(const float2*)(Arow + kn + klo);
            }
            if (kk + 64 < NN)
                __builtin_prefetch(Arow + kk + 64, 0, 0);
        }

        // deg on the matrix pipe: D += A_hi@1 + A_lo@1.
        dacc = wmma_bf16(ah, onesu, dacc);
        dacc = wmma_bf16(al, onesu, dacc);

        // B-fragment double buffer: load tile c+1 while computing tile c.
        v8u bh = *(const v8u*)&sXh[(bfh * 128 + (l & 15)) * 8];
        v8u bl = *(const v8u*)&sXl[(bfh * 128 + (l & 15)) * 8];
        #pragma unroll
        for (int c = 0; c < 8; ++c) {
            v8u bhn, bln;
            if (c < 7) {
                int nb = (c + 1) * 16 + (l & 15);
                bhn = *(const v8u*)&sXh[(bfh * 128 + nb) * 8];
                bln = *(const v8u*)&sXl[(bfh * 128 + nb) * 8];
            }
            acc[c] = wmma_bf16(ah, bh, acc[c]);   // hi*hi
            acc[c] = wmma_bf16(al, bh, acc[c]);   // lo*hi
            acc[c] = wmma_bf16(ah, bl, acc[c]);   // hi*lo
            if (c < 7) { bh = bhn; bl = bln; }
        }
    }

    // ------------- Epilogue 1: H = (acc + X)/deg -> LDS (A-operand layout) ----
    __syncthreads();                      // done reading sX before aliasing
    unsigned short* sH = (unsigned short*)smem;   // 32768 halves
    const int hbase = w * 4096;           // per-wave: hi [0,2048), lo [2048,4096)
    #pragma unroll
    for (int c = 0; c < 8; ++c) {
        int n   = c * 16 + (l & 15);      // feature = K of GEMM2
        int kt  = n >> 5, kq = n & 31;
        int ks2 = (kq >> 3) & 1;
        int r2  = ((kq & 7) >> 1) + ((kq >> 4) << 2);
        int hf  = kq & 1;
        #pragma unroll
        for (int r = 0; r < 8; ++r) {
            int M = r + 8 * (l >> 4);     // C-layout row
            int flatRow = flatRow0 + w * 16 + M;
            float xadd = X[(size_t)flatRow * IN_F + n];
            float deg  = dacc[r] + 1.0f;
            float h    = (acc[c][r] + xadd) / deg;
            unsigned int hh = f2bf(h);
            unsigned int hl = f2bf(h - bf2f(hh));
            int idx = (((M * 4 + kt) * 2 + ks2) * 8 + r2) * 2 + hf;
            sH[hbase + idx]        = (unsigned short)hh;
            sH[hbase + 2048 + idx] = (unsigned short)hl;
        }
    }
    __syncthreads();

    // ---------------- Phase 2: out = relu(H @ W), bf16 hi/lo x3 --------------
    // W fragments come pre-packed from ws (L2-resident, zero conversion VALU).
    v8f acc2[8];
    #pragma unroll
    for (int c = 0; c < 8; ++c)
        #pragma unroll
        for (int i = 0; i < 8; ++i) acc2[c][i] = 0.0f;

    const int mA2 = l & 15;
    const int ksA = l >> 4;
    const int kh2 = l >> 4;
    #pragma unroll
    for (int kt = 0; kt < 4; ++kt) {
        int dbase = w * 2048 + ((mA2 * 4 + kt) * 2 + ksA) * 8;
        v8u ah2 = *(const v8u*)&smem[dbase];
        v8u al2 = *(const v8u*)&smem[dbase + 1024];
        #pragma unroll
        for (int c = 0; c < 8; ++c) {
            int nb = c * 16 + (l & 15);
            int widx = ((kt * 2 + kh2) * 128 + nb) * 8;
            v8u bwh = *(const v8u*)&wsW[widx];         // 2x global_load_b128
            v8u bwl = *(const v8u*)&wsW[widx + 8192];
            acc2[c] = wmma_bf16(ah2, bwh, acc2[c]);
            acc2[c] = wmma_bf16(al2, bwh, acc2[c]);
            acc2[c] = wmma_bf16(ah2, bwl, acc2[c]);
        }
    }

    #pragma unroll
    for (int c = 0; c < 8; ++c) {
        int n = c * 16 + (l & 15);
        #pragma unroll
        for (int r = 0; r < 8; ++r) {
            int flatRow = flatRow0 + w * 16 + r + 8 * (l >> 4);
            out[(size_t)flatRow * OUT_F + n] = fmaxf(acc2[c][r], 0.0f);
        }
    }
}

extern "C" void kernel_launch(void* const* d_in, const int* in_sizes, int n_in,
                              void* d_out, int out_size, void* d_ws, size_t ws_size,
                              hipStream_t stream) {
    const float* A = (const float*)d_in[0];   // relation_matrix [B,N,N]
    const float* X = (const float*)d_in[1];   // inputs          [B,N,128]
    const float* W = (const float*)d_in[2];   // w               [128,128]
    float* out = (float*)d_out;               // [B,N,128]
    unsigned int* wsW = (unsigned int*)d_ws;  // 64 KB packed-W fragments
    (void)in_sizes; (void)n_in; (void)out_size; (void)ws_size;

    pack_w_kernel<<<dim3(1), dim3(256), 0, stream>>>(W, wsW);
    graphconv_fused<<<dim3(NB * NN / 128), dim3(256), 0, stream>>>(A, X, wsW, out);
}
